// VarianceAdaptor_51943334478373
// MI455X (gfx1250) — matrix operand order repaired and checked by reading, not compiled, not run on hardware
//
#include <hip/hip_runtime.h>

typedef __bf16 bf16;
typedef __attribute__((ext_vector_type(16))) __bf16 v16bf;
typedef __attribute__((ext_vector_type(8)))  float   v8f;

#define BT    128   // time steps per batch item
#define DC    256   // channels
#define XROWD 132   // dwords per Xs row (264 bf16; 528B: 16B-aligned, optimal bank tiling)
#define BROWD 132   // dwords per Bs row
#define MAXLEN 80

#if defined(__has_builtin) && __has_builtin(__builtin_amdgcn_sched_group_barrier)
#define SGB(mask, size, id) __builtin_amdgcn_sched_group_barrier(mask, size, id)
#else
#define SGB(mask, size, id)
#endif

union Frag   { uint4 q[2]; unsigned int u[8]; v16bf v; };
union PackCv { __bf16 h[2]; unsigned int u; unsigned short s[2]; };

__device__ inline unsigned int pack2(float a, float b) {
  PackCv c; c.h[0] = (__bf16)a; c.h[1] = (__bf16)b; return c.u;
}
__device__ inline unsigned short pack1(float a) {
  PackCv c; c.u = 0u; c.h[0] = (__bf16)a; return c.s[0];
}

// ---------------------------------------------------------------------------
// Repack: convert conv weights [3,256,256] f32 -> bf16 (same element order).
// [tap][cin][cout] order is already [chunk][k][n]-major for 32-deep K chunks.
// ---------------------------------------------------------------------------
__global__ __launch_bounds__(256) void va_repack(
    const float* __restrict__ w1, const float* __restrict__ w2,
    bf16* __restrict__ dst)
{
  int g = blockIdx.x * 256 + threadIdx.x;       // 0 .. 393215
  int cw = (g >= 3 * DC * DC) ? 1 : 0;
  int r  = g - cw * (3 * DC * DC);
  const float* src = cw ? w2 : w1;
  dst[g] = (bf16)src[r];
}

// staged weight-chunk global load (16 dwords/thread, coalesced, L2-resident)
__device__ inline void wload(uint4* wr, const bf16* __restrict__ wpack,
                             int conv, int kc, int k, int seg) {
  const uint4* gp = (const uint4*)(wpack + ((size_t)(conv * 24 + kc) * 32 + k) * DC + seg * 32);
  wr[0] = gp[0]; wr[1] = gp[1]; wr[2] = gp[2]; wr[3] = gp[3];
}
__device__ inline void wstore(unsigned int* bs, const uint4* wr, int k, int seg) {
  uint4* dp = (uint4*)(bs + k * BROWD + seg * 16);
  dp[0] = wr[0]; dp[1] = wr[1]; dp[2] = wr[2]; dp[3] = wr[3];
}
__device__ inline void loadB(Frag& f, const unsigned short* base, int nt) {
  const unsigned short* p = base + nt * 16;
  #pragma unroll
  for (int j = 0; j < 8; ++j)
    f.u[j] = (unsigned int)p[(2 * j) * (2 * BROWD)] | ((unsigned int)p[(2 * j + 1) * (2 * BROWD)] << 16);
}

// ---------------------------------------------------------------------------
// Main: one block per batch item; 8 waves of 32.
//   conv1 (WMMA) -> ReLU -> LN -> conv2 (WMMA) -> ReLU -> LN -> dense
//   -> durations -> cumsum -> length-regulated gather of f32 encoder rows.
// Pipelined: weight chunk kc+1 fetched from L2 while chunk kc computes;
// B fragments double-buffered; sched_group_barrier pins the DS/WMMA
// interleave so LDS latency hides behind v_wmma issue.
// ---------------------------------------------------------------------------
__global__ __launch_bounds__(256) void va_main(
    const float* __restrict__ enc, const bf16* __restrict__ wpack,
    const float* __restrict__ b1,  const float* __restrict__ b2,
    const float* __restrict__ gam, const float* __restrict__ bet,
    const float* __restrict__ dw,  const float* __restrict__ db,
    float* __restrict__ out)
{
  __shared__ __align__(16) unsigned int XsU[BT * XROWD];      // activations (bf16 pairs)
  __shared__ __align__(16) unsigned int BsU[2][32 * BROWD];   // weight chunks, double-buffered
  __shared__ int cumS[BT];

  const int tid  = threadIdx.x;
  const int lane = tid & 31;
  const int w    = tid >> 5;        // wave id = M-tile id (0..7)
  const int b    = blockIdx.x;
  const int nn   = lane & 15;
  const int kh   = lane >> 4;
  const int sk   = tid >> 3;        // staging: k row   (0..31)
  const int sseg = tid & 7;         // staging: segment (0..7)
  unsigned short* XsH = (unsigned short*)XsU;

  // ---- Phase 0: stage encoder block as bf16 into LDS ----
  {
    const float2* src = (const float2*)(enc + (size_t)b * BT * DC);
    #pragma unroll
    for (int j = 0; j < (BT * DC / 2) / 256; ++j) {
      int p   = tid + j * 256;        // pair index
      int row = p >> 7;               // 128 pairs per row
      int c2  = p & 127;
      float2 v = src[p];
      XsU[row * XROWD + c2] = pack2(v.x, v.y);
    }
  }

  for (int conv = 0; conv < 2; ++conv) {
    v8f acc[16];
    #pragma unroll
    for (int nt = 0; nt < 16; ++nt) acc[nt] = (v8f){0.f,0.f,0.f,0.f,0.f,0.f,0.f,0.f};

    const float* biasp = conv ? b2 : b1;

    // pipeline prologue: stage chunk 0 into buffer 0 (also fences phase-0 Xs writes)
    {
      uint4 wr[4];
      wload(wr, wpack, conv, 0, sk, sseg);
      wstore(BsU[0], wr, sk, sseg);
    }
    __syncthreads();

    #pragma unroll 1
    for (int kc = 0; kc < 24; ++kc) {
      const int buf = kc & 1;
      const int tap = kc >> 3;
      const int cb  = (kc & 7) * 32;

      // issue global fetch of next weight chunk (hides L2 latency behind WMMAs)
      uint4 wr[4];
      if (kc < 23) wload(wr, wpack, conv, kc + 1, sk, sseg);

      // A fragment: 16x32 bf16, lane=M, VGPR pairs=K (contiguous k = 16*kh + s, same order as B)
      Frag a;
      const int mrow = w * 16 + nn + tap - 1;   // implicit im2col with zero pad
      if (mrow >= 0 && mrow < BT) {
        const uint4* p = (const uint4*)(XsU + mrow * XROWD + (cb >> 1) + 8 * kh);
        a.q[0] = p[0];       // K 16*kh+0..7
        a.q[1] = p[1];       // K 16*kh+8..15
      } else {
        a.q[0] = (uint4){0u,0u,0u,0u};
        a.q[1] = (uint4){0u,0u,0u,0u};
      }

      // 16 N-tiles, B fragments double-buffered: loads for nt+1 overlap wmma nt
      const unsigned short* bbase = (const unsigned short*)(BsU[buf] + kh * 16 * BROWD) + nn;
      Frag fb[2];
      loadB(fb[0], bbase, 0);
      #pragma unroll
      for (int nt = 0; nt < 16; ++nt) {
        if (nt < 15) loadB(fb[(nt + 1) & 1], bbase, nt + 1);
        acc[nt] = __builtin_amdgcn_wmma_f32_16x16x32_bf16(
            false, a.v, false, fb[nt & 1].v, (short)0, acc[nt], false, false);
      }
      // Pin the software pipeline: [2 DS-reads (nt=0)],
      // then 15x [2 DS-reads (nt+1)][1 WMMA (nt)], then the final WMMA.
      SGB(0x100, 2, 0);                  // B-frag loads nt=0
      #pragma unroll
      for (int i = 0; i < 15; ++i) {
        SGB(0x100, 2, 0);                // B-frag loads nt=i+1
        SGB(0x008, 1, 0);                // wmma nt=i
      }
      SGB(0x008, 1, 0);                  // wmma nt=15

      // commit next weight chunk to the other LDS buffer
      if (kc < 23) wstore(BsU[buf ^ 1], wr, sk, sseg);
      __syncthreads();
    }

    // ---- epilogue: bias + ReLU ----
    #pragma unroll
    for (int nt = 0; nt < 16; ++nt) {
      float bb = biasp[nt * 16 + nn];
      #pragma unroll
      for (int r = 0; r < 8; ++r) {
        float x = acc[nt][r] + bb;
        acc[nt][r] = x > 0.f ? x : 0.f;
      }
    }
    // ---- LayerNorm stats: row m = w*16 + r + 8*kh lives in one lane-half ----
    float mur[8], rsr[8];
    #pragma unroll
    for (int r = 0; r < 8; ++r) {
      float s = 0.f, s2 = 0.f;
      #pragma unroll
      for (int nt = 0; nt < 16; ++nt) { float x = acc[nt][r]; s += x; s2 += x * x; }
      #pragma unroll
      for (int off = 1; off < 16; off <<= 1) {
        s  += __shfl_xor(s,  off, 16);
        s2 += __shfl_xor(s2, off, 16);
      }
      float mu  = s  * (1.f / 256.f);
      float var = s2 * (1.f / 256.f) - mu * mu;
      mur[r] = mu;
      rsr[r] = rsqrtf(var + 1e-6f);
    }

    if (conv == 0) {
      // normalize and write bf16 back to Xs (conv2 input)
      #pragma unroll
      for (int nt = 0; nt < 16; ++nt) {
        int n = nt * 16 + nn;
        float g = gam[n], be = bet[n];
        #pragma unroll
        for (int r = 0; r < 8; ++r) {
          float y = (acc[nt][r] - mur[r]) * rsr[r] * g + be;
          int row = w * 16 + r + 8 * kh;
          XsH[row * (2 * XROWD) + n] = pack1(y);
        }
      }
    } else {
      // normalize + dense head -> per-row duration
      float dp[8] = {0.f,0.f,0.f,0.f,0.f,0.f,0.f,0.f};
      #pragma unroll
      for (int nt = 0; nt < 16; ++nt) {
        int n = nt * 16 + nn;
        float g = gam[n], be = bet[n], wd = dw[n];
        #pragma unroll
        for (int r = 0; r < 8; ++r) {
          float y = (acc[nt][r] - mur[r]) * rsr[r] * g + be;
          dp[r] += y * wd;
        }
      }
      float bd = db[0];
      #pragma unroll
      for (int r = 0; r < 8; ++r) {
        float d = dp[r];
        #pragma unroll
        for (int off = 1; off < 16; off <<= 1) d += __shfl_xor(d, off, 16);
        d = d + bd;
        d = fminf(fmaxf(d, 0.f), 75.f);
        int dur = (int)rintf(d);   // round-half-even, matches jnp.round
        if (nn == 0) cumS[w * 16 + r + 8 * kh] = dur;
      }
    }
    __syncthreads();
  }

  // ---- cumsum (tiny, serial) ----
  if (tid == 0) {
    int run = 0;
    for (int t = 0; t < BT; ++t) { run += cumS[t]; cumS[t] = run; }
  }
  __syncthreads();

  // ---- length regulation: gather f32 encoder rows ----
  const int total = cumS[BT - 1];
  const float4* encb = (const float4*)(enc + (size_t)b * BT * DC);
  float4* outb = (float4*)(out + (size_t)b * MAXLEN * DC);
  for (int p = w; p < MAXLEN; p += 8) {
    int cnt = 0;
    int t0  = lane * 4;
    #pragma unroll
    for (int j = 0; j < 4; ++j) cnt += (p >= cumS[t0 + j]) ? 1 : 0;
    #pragma unroll
    for (int off = 16; off >= 1; off >>= 1) cnt += __shfl_xor(cnt, off, 32);
    int  idx   = cnt < (BT - 1) ? cnt : (BT - 1);
    bool valid = p < total;
    const float4* src = encb + (size_t)idx * (DC / 4);
    float4 z = {0.f, 0.f, 0.f, 0.f};
    float4 v0 = valid ? src[lane]      : z;
    float4 v1 = valid ? src[lane + 32] : z;
    outb[(size_t)p * (DC / 4) + lane]      = v0;
    outb[(size_t)p * (DC / 4) + lane + 32] = v1;
  }
}

extern "C" void kernel_launch(void* const* d_in, const int* in_sizes, int n_in,
                              void* d_out, int out_size, void* d_ws, size_t ws_size,
                              hipStream_t stream)
{
  const float* enc = (const float*)d_in[0];
  const float* w1  = (const float*)d_in[1];
  const float* b1  = (const float*)d_in[2];
  const float* w2  = (const float*)d_in[3];
  const float* b2  = (const float*)d_in[4];
  const float* gam = (const float*)d_in[5];
  const float* bet = (const float*)d_in[6];
  const float* dw  = (const float*)d_in[7];
  const float* db  = (const float*)d_in[8];
  (void)in_sizes; (void)n_in; (void)out_size; (void)ws_size;

  bf16* wpack = (bf16*)d_ws;   // 2 * 196608 bf16 = 768 KB
  va_repack<<<(2 * 3 * 256 * 256) / 256, 256, 0, stream>>>(w1, w2, wpack);
  va_main<<<2048, 256, 0, stream>>>(enc, wpack, b1, b2, gam, bet, dw, db, (float*)d_out);
}
